// SimpleDecoder_82514911691464
// MI455X (gfx1250) — compile-verified, matrix operand order checked by reading, not run on hardware
//
#include <hip/hip_runtime.h>
#include <cstdint>
#include <cstddef>

#define DM 256
#define NH 8
#define DK 32
#define NL 4
#define NVOC 1000
#define BS 8
#define SQ 2048
#define NTOK (BS * SQ)

typedef __attribute__((ext_vector_type(16))) __bf16 v16bf;
typedef __attribute__((ext_vector_type(8)))  float  v8f;
typedef __attribute__((ext_vector_type(8)))  unsigned short v8u16;

// ---- optional CDNA5 async global->LDS path (guarded so compile never breaks) ----
#if defined(__has_builtin)
#if __has_builtin(__builtin_amdgcn_global_load_async_to_lds_b128) && \
    __has_builtin(__builtin_amdgcn_s_wait_asynccnt)
#define USE_ASYNC_LDS 1
#endif
#endif
#ifndef USE_ASYNC_LDS
#define USE_ASYNC_LDS 0
#endif

#if USE_ASYNC_LDS
// signature (from hipcc diagnostic): (v4i AS1* src, v4i AS3* dst, Imm offset, Imm cpol)
typedef int v4i __attribute__((vector_size(16)));
typedef __attribute__((address_space(1))) v4i* gv4p;
typedef __attribute__((address_space(3))) v4i* lv4p;
#endif

union FragU {
  v16bf bf;
  v8u16 h[2];
  unsigned short s[16];
};

__device__ __forceinline__ unsigned short f2bf(float f) {
  unsigned int u = __float_as_uint(f);
  u += 0x7FFFu + ((u >> 16) & 1u);   // round-to-nearest-even
  return (unsigned short)(u >> 16);
}

// Load a 16x32 bf16 A/B fragment from row-major memory (ld elems, 16B-aligned rows).
// ISA layout: lanes 0-15 -> row M=lane, K = k0+{0..7, 16..23}; lanes 16-31 -> K = k0+{8..15, 24..31}.
__device__ __forceinline__ v16bf load_frag(const unsigned short* p, int ld, int r0, int k0) {
  int lane = threadIdx.x & 31;
  size_t row = (size_t)(r0 + (lane & 15));
  int kk = k0 + ((lane >> 4) << 3);
  const unsigned short* base = p + row * (size_t)ld + kk;
  FragU u;
  u.h[0] = *(const v8u16*)(base);
  u.h[1] = *(const v8u16*)(base + 16);
  return u.bf;
}

__device__ __forceinline__ v16bf load_frag_guard(const unsigned short* p, int ld, int r0,
                                                 int k0, int rmax) {
  int lane = threadIdx.x & 31;
  int row = r0 + (lane & 15);
  int kk = k0 + ((lane >> 4) << 3);
  FragU u;
  if (row < rmax) {
    const unsigned short* base = p + (size_t)row * (size_t)ld + kk;
    u.h[0] = *(const v8u16*)(base);
    u.h[1] = *(const v8u16*)(base + 16);
  } else {
#pragma unroll
    for (int i = 0; i < 16; ++i) u.s[i] = 0;
  }
  return u.bf;
}

// ---------------- elementwise kernels ----------------

__global__ void conv_bf16_kernel(const float* __restrict__ src,
                                 unsigned short* __restrict__ dst, int n) {
  int i = blockIdx.x * blockDim.x + threadIdx.x;
  if (i < n) dst[i] = f2bf(src[i]);
}

__global__ void embed_gather_kernel(const int* __restrict__ tok,
                                    const float* __restrict__ emb,
                                    unsigned short* __restrict__ xb) {
  int i = blockIdx.x * blockDim.x + threadIdx.x;   // over NTOK*DM
  int n = i >> 8;
  int d = i & (DM - 1);
  int t = tok[n];
  xb[i] = f2bf(emb[(size_t)t * DM + d]);
}

// ---------------- GEMM: Y[N,256] = X[N,256] @ W[256,256]^T (bf16 in, f32 acc) ----------------
// Block: 128 threads = 4 waves; wave w does rows [bx*64 + w*16, +16), cols [by*64, +64).
// The 64x256 W tile (contiguous 32KB) is staged in LDS via async-to-LDS;
// B fragments come from LDS, A fragments are register-pipelined one k-step ahead.

__global__ __launch_bounds__(128)
void gemm_bf16_kernel(const unsigned short* __restrict__ X,
                      const unsigned short* __restrict__ W,
                      unsigned short* __restrict__ Yb,
                      float* __restrict__ Yf, int writeF) {
  __shared__ unsigned short Wl[64 * DM];   // 32 KB
  int tid = threadIdx.x;
  int wave = tid >> 5;
  int lane = tid & 31;
  int r0 = blockIdx.x * 64 + wave * 16;
  int n0 = blockIdx.y * 64;

  // stage W tile: rows [n0, n0+64) x all K -> one contiguous 64*256 element block
  {
    const unsigned short* wsrc = W + (size_t)n0 * DM;
#if USE_ASYNC_LDS
#pragma unroll
    for (int i = 0; i < 16; ++i) {
      int c = (i * 128 + tid) * 8;
      __builtin_amdgcn_global_load_async_to_lds_b128(
          (gv4p)(wsrc + c), (lv4p)(Wl + c), 0, 0);
    }
    __builtin_amdgcn_s_wait_asynccnt(0);
#else
#pragma unroll
    for (int i = 0; i < 16; ++i) {
      int c = (i * 128 + tid) * 8;
      *(v8u16*)(Wl + c) = *(const v8u16*)(wsrc + c);
    }
#endif
  }
  __syncthreads();

  v8f acc[4] = {};
  v16bf a = load_frag(X, DM, r0, 0);
#pragma unroll
  for (int ks = 0; ks < 8; ++ks) {
    int k = ks * 32;
    v16bf b0 = load_frag(Wl, DM, 0,  k);
    v16bf b1 = load_frag(Wl, DM, 16, k);
    v16bf b2 = load_frag(Wl, DM, 32, k);
    v16bf b3 = load_frag(Wl, DM, 48, k);
    v16bf an = a;
    if (ks < 7) an = load_frag(X, DM, r0, k + 32);   // compile-time branch (unrolled)
    acc[0] = __builtin_amdgcn_wmma_f32_16x16x32_bf16(false, a, false, b0, (short)0, acc[0], false, false);
    acc[1] = __builtin_amdgcn_wmma_f32_16x16x32_bf16(false, a, false, b1, (short)0, acc[1], false, false);
    acc[2] = __builtin_amdgcn_wmma_f32_16x16x32_bf16(false, a, false, b2, (short)0, acc[2], false, false);
    acc[3] = __builtin_amdgcn_wmma_f32_16x16x32_bf16(false, a, false, b3, (short)0, acc[3], false, false);
    a = an;
  }

  int mb = (lane >> 4) << 3;
  int nn = lane & 15;
#pragma unroll
  for (int t = 0; t < 4; ++t) {
#pragma unroll
    for (int r = 0; r < 8; ++r) {
      size_t idx = (size_t)(r0 + mb + r) * DM + (n0 + t * 16 + nn);
      float v = acc[t][r];
      Yb[idx] = f2bf(v);
      if (writeF) Yf[idx] = v;
    }
  }
}

// ---------------- vocab head: Y[N,1000] = Xn @ headW^T + b ----------------

__global__ __launch_bounds__(128)
void head_gemm_kernel(const unsigned short* __restrict__ X,
                      const unsigned short* __restrict__ W,
                      const float* __restrict__ bias,
                      float* __restrict__ Y) {
  int wave = threadIdx.x >> 5;
  int lane = threadIdx.x & 31;
  int r0 = blockIdx.x * 64 + wave * 16;
  int n0 = blockIdx.y * 64;
  v8f acc[4] = {};
  for (int k = 0; k < DM; k += 32) {
    v16bf a = load_frag(X, DM, r0, k);
#pragma unroll
    for (int t = 0; t < 4; ++t) {
      v16bf b = load_frag_guard(W, DM, n0 + t * 16, k, NVOC);
      acc[t] = __builtin_amdgcn_wmma_f32_16x16x32_bf16(false, a, false, b,
                                                       (short)0, acc[t], false, false);
    }
  }
  int mb = (lane >> 4) << 3;
  int nn = lane & 15;
#pragma unroll
  for (int t = 0; t < 4; ++t) {
    int n = n0 + t * 16 + nn;
    if (n < NVOC) {
      float bv = bias[n];
#pragma unroll
      for (int r = 0; r < 8; ++r) {
        Y[(size_t)(r0 + mb + r) * NVOC + n] = acc[t][r] + bv;
      }
    }
  }
}

// ---------------- streaming (flash) attention, one wave per (b,h,16-row q tile) ----------------

#define VT_LD 40

__global__ __launch_bounds__(32)
void attn_kernel(const unsigned short* __restrict__ Q,
                 const unsigned short* __restrict__ K,
                 const unsigned short* __restrict__ V,
                 unsigned short* __restrict__ O) {
  __shared__ unsigned short Vt[32 * VT_LD];   // V block transposed: Vt[d][k]
  __shared__ unsigned short Pl[16 * 32];      // P tile in row-major for A-frag reload
  int lane = threadIdx.x;
  int qt = blockIdx.x;
  int h  = blockIdx.y;
  int b  = blockIdx.z;
  int rowBase = b * SQ;
  int q0 = rowBase + qt * 16;
  int c0 = h * DK;
  const float scale = 0.17677669529663687f;   // 1/sqrt(32)

  v16bf aQ = load_frag(Q, DM, q0, c0);
  v8f o0 = {}, o1 = {};
  float rm[8], rl[8];
#pragma unroll
  for (int r = 0; r < 8; ++r) { rm[r] = -3.0e38f; rl[r] = 0.0f; }

  for (int kb = 0; kb < SQ; kb += 32) {
    if (kb + 32 < SQ) {
      __builtin_prefetch(K + (size_t)(rowBase + kb + 32 + lane) * DM + c0, 0, 1);
      __builtin_prefetch(V + (size_t)(rowBase + kb + 32 + lane) * DM + c0, 0, 1);
    }
    // stage V block transposed into LDS: lane loads V row (kb+lane), scatters 32 d's
    {
      const unsigned short* vrow = V + (size_t)(rowBase + kb + lane) * DM + c0;
      v8u16 cc[4];
      cc[0] = *(const v8u16*)(vrow);
      cc[1] = *(const v8u16*)(vrow + 8);
      cc[2] = *(const v8u16*)(vrow + 16);
      cc[3] = *(const v8u16*)(vrow + 24);
#pragma unroll
      for (int j = 0; j < 32; ++j) Vt[j * VT_LD + lane] = cc[j >> 3][j & 7];
    }
    // scores S = Q K^T  (16 x 32, two 16x16 WMMA tiles)
    v16bf bK0 = load_frag(K, DM, rowBase + kb, c0);
    v16bf bK1 = load_frag(K, DM, rowBase + kb + 16, c0);
    v8f z = {};
    v8f s0 = __builtin_amdgcn_wmma_f32_16x16x32_bf16(false, aQ, false, bK0, (short)0, z, false, false);
    v8f s1 = __builtin_amdgcn_wmma_f32_16x16x32_bf16(false, aQ, false, bK1, (short)0, z, false, false);

    // online softmax row update (row = M lives across 16 lanes in accumulator layout)
    float alpha[8];
#pragma unroll
    for (int r = 0; r < 8; ++r) {
      float x0 = s0[r] * scale;
      float x1 = s1[r] * scale;
      float t = fmaxf(x0, x1);
      t = fmaxf(t, __shfl_xor(t, 1, 32));
      t = fmaxf(t, __shfl_xor(t, 2, 32));
      t = fmaxf(t, __shfl_xor(t, 4, 32));
      t = fmaxf(t, __shfl_xor(t, 8, 32));
      float mnew = fmaxf(rm[r], t);
      alpha[r] = __expf(rm[r] - mnew);
      rm[r] = mnew;
      float p0 = __expf(x0 - mnew);
      float p1 = __expf(x1 - mnew);
      float rs = p0 + p1;
      rs += __shfl_xor(rs, 1, 32);
      rs += __shfl_xor(rs, 2, 32);
      rs += __shfl_xor(rs, 4, 32);
      rs += __shfl_xor(rs, 8, 32);
      rl[r] = rl[r] * alpha[r] + rs;
      s0[r] = p0;
      s1[r] = p1;
    }
    // re-layout P (acc layout -> A-frag layout) via LDS
    {
      int mb = (lane >> 4) << 3;
      int nn = lane & 15;
#pragma unroll
      for (int r = 0; r < 8; ++r) {
        Pl[(mb + r) * 32 + nn]      = f2bf(s0[r]);
        Pl[(mb + r) * 32 + nn + 16] = f2bf(s1[r]);
      }
    }
    __syncthreads();
    v16bf aP  = load_frag(Pl, 32, 0, 0);
    v16bf bV0 = load_frag(Vt, VT_LD, 0, 0);
    v16bf bV1 = load_frag(Vt, VT_LD, 16, 0);
#pragma unroll
    for (int r = 0; r < 8; ++r) { o0[r] *= alpha[r]; o1[r] *= alpha[r]; }
    o0 = __builtin_amdgcn_wmma_f32_16x16x32_bf16(false, aP, false, bV0, (short)0, o0, false, false);
    o1 = __builtin_amdgcn_wmma_f32_16x16x32_bf16(false, aP, false, bV1, (short)0, o1, false, false);
    __syncthreads();
  }

  int mb = (lane >> 4) << 3;
  int nn = lane & 15;
#pragma unroll
  for (int r = 0; r < 8; ++r) {
    float inv = 1.0f / rl[r];
    size_t row = (size_t)(q0 + mb + r) * DM;
    O[row + c0 + nn]      = f2bf(o0[r] * inv);
    O[row + c0 + 16 + nn] = f2bf(o1[r] * inv);
  }
}

// ---------------- LayerNorm (wave per token) -> bf16 ----------------

__global__ __launch_bounds__(128)
void ln_kernel(const float* __restrict__ X, const float* __restrict__ gamma,
               const float* __restrict__ beta, unsigned short* __restrict__ Xn) {
  int wave = threadIdx.x >> 5;
  int lane = threadIdx.x & 31;
  int tokIdx = blockIdx.x * 4 + wave;
  const float* x = X + (size_t)tokIdx * DM;
  float v[8];
  float s = 0.0f;
#pragma unroll
  for (int j = 0; j < 8; ++j) { v[j] = x[j * 32 + lane]; s += v[j]; }
#pragma unroll
  for (int m = 16; m >= 1; m >>= 1) s += __shfl_xor(s, m, 32);
  float mu = s * (1.0f / DM);
  float q = 0.0f;
#pragma unroll
  for (int j = 0; j < 8; ++j) { float d = v[j] - mu; q += d * d; }
#pragma unroll
  for (int m = 16; m >= 1; m >>= 1) q += __shfl_xor(q, m, 32);
  float rstd = rsqrtf(q * (1.0f / DM) + 1e-5f);
  unsigned short* out = Xn + (size_t)tokIdx * DM;
#pragma unroll
  for (int j = 0; j < 8; ++j) {
    int d = j * 32 + lane;
    out[d] = f2bf((v[j] - mu) * rstd * gamma[d] + beta[d]);
  }
}

// ---------------- host-side orchestration ----------------

extern "C" void kernel_launch(void* const* d_in, const int* in_sizes, int n_in,
                              void* d_out, int out_size, void* d_ws, size_t ws_size,
                              hipStream_t stream) {
  (void)in_sizes; (void)n_in; (void)out_size; (void)ws_size;
  const int*   tokens = (const int*)d_in[0];
  const float* embed  = (const float*)d_in[1];
  const float* Wq     = (const float*)d_in[2];
  const float* Wk     = (const float*)d_in[3];
  const float* Wv     = (const float*)d_in[4];
  const float* Wo     = (const float*)d_in[5];
  const float* gamma  = (const float*)d_in[6];
  const float* beta   = (const float*)d_in[7];
  const float* headW  = (const float*)d_in[8];
  const float* headb  = (const float*)d_in[9];
  float* out = (float*)d_out;

  char* ws = (char*)d_ws;
  size_t off = 0;
  auto alloc = [&](size_t bytes) -> void* {
    void* p = ws + off;
    off = (off + bytes + 255) & ~(size_t)255;
    return p;
  };

  const size_t WMAT = (size_t)NL * DM * DM;          // 262144 per weight type
  const size_t ACT  = (size_t)NTOK * DM;             // 4.19M elements

  unsigned short* wqb = (unsigned short*)alloc(WMAT * 2);
  unsigned short* wkb = (unsigned short*)alloc(WMAT * 2);
  unsigned short* wvb = (unsigned short*)alloc(WMAT * 2);
  unsigned short* wob = (unsigned short*)alloc(WMAT * 2);
  unsigned short* hwb = (unsigned short*)alloc((size_t)NVOC * DM * 2);
  unsigned short* xbA = (unsigned short*)alloc(ACT * 2);
  unsigned short* xbB = (unsigned short*)alloc(ACT * 2);
  unsigned short* qb  = (unsigned short*)alloc(ACT * 2);
  unsigned short* kbf = (unsigned short*)alloc(ACT * 2);
  unsigned short* vb  = (unsigned short*)alloc(ACT * 2);
  unsigned short* ab  = (unsigned short*)alloc(ACT * 2);
  unsigned short* xn  = (unsigned short*)alloc(ACT * 2);
  float*          x32 = (float*)alloc(ACT * 4);

  // weights -> bf16 (deterministic each call)
  conv_bf16_kernel<<<(int)((WMAT + 255) / 256), 256, 0, stream>>>(Wq, wqb, (int)WMAT);
  conv_bf16_kernel<<<(int)((WMAT + 255) / 256), 256, 0, stream>>>(Wk, wkb, (int)WMAT);
  conv_bf16_kernel<<<(int)((WMAT + 255) / 256), 256, 0, stream>>>(Wv, wvb, (int)WMAT);
  conv_bf16_kernel<<<(int)((WMAT + 255) / 256), 256, 0, stream>>>(Wo, wob, (int)WMAT);
  conv_bf16_kernel<<<(NVOC * DM + 255) / 256, 256, 0, stream>>>(headW, hwb, NVOC * DM);

  // x = embed[tokens] (bf16)
  embed_gather_kernel<<<NTOK * DM / 256, 256, 0, stream>>>(tokens, embed, xbA);

  unsigned short* cur = xbA;
  unsigned short* nxt = xbB;
  dim3 ggrid(NTOK / 64, DM / 64);      // (256, 4)
  dim3 agrid(SQ / 16, NH, BS);         // (128, 8, 8)
  for (int l = 0; l < NL; ++l) {
    const unsigned short* wq = wqb + (size_t)l * DM * DM;
    const unsigned short* wk = wkb + (size_t)l * DM * DM;
    const unsigned short* wv = wvb + (size_t)l * DM * DM;
    const unsigned short* wo = wob + (size_t)l * DM * DM;
    gemm_bf16_kernel<<<ggrid, 128, 0, stream>>>(cur, wq, qb,  nullptr, 0);
    gemm_bf16_kernel<<<ggrid, 128, 0, stream>>>(cur, wk, kbf, nullptr, 0);
    gemm_bf16_kernel<<<ggrid, 128, 0, stream>>>(cur, wv, vb,  nullptr, 0);
    attn_kernel<<<agrid, 32, 0, stream>>>(qb, kbf, vb, ab);
    gemm_bf16_kernel<<<ggrid, 128, 0, stream>>>(ab, wo, nxt, x32, 1);
    unsigned short* t = cur; cur = nxt; nxt = t;
  }

  ln_kernel<<<NTOK / 4, 128, 0, stream>>>(x32, gamma, beta, xn);
  head_gemm_kernel<<<dim3(NTOK / 64, 16), 128, 0, stream>>>(xn, hwb, headb, out);
}
